// SelfAttentionLayer_72129680769842
// MI455X (gfx1250) — matrix operand order repaired and checked by reading, not compiled, not run on hardware
//
#include <hip/hip_runtime.h>

// ---------------------------------------------------------------------------
// Self-attention for MI455X (gfx1250, wave32, WMMA f16 16x16x32, f32 accum)
//   B=4, S=2048, D=1024, H=16, Dh=64
// Kernel 1: QKV projection GEMM (fp32 in -> f16 head-split Q/K/V in d_ws)
// Kernel 2: flash attention, 64-key tiles, 16 q-rows/wave (softmax VALU
//           overhead per key cut 4x vs 32-key tiles; WMMA ratio unchanged)
// ---------------------------------------------------------------------------

typedef __attribute__((ext_vector_type(16))) _Float16 v16h;
typedef __attribute__((ext_vector_type(8)))  float    v8f;
typedef _Float16 h16;

#define BATCH 4
#define SEQ   2048
#define DMODEL 1024
#define HEADS 16
#define DHEAD 64

__device__ __forceinline__ v8f wmma_f16(v16h a, v16h b, v8f c) {
    // D = A(16x32 f16) * B(32x16 f16) + C(16x16 f32)
    return __builtin_amdgcn_wmma_f32_16x16x32_f16(
        /*neg_a=*/false, a, /*neg_b=*/false, b,
        /*c_mod=*/(short)0, c, /*reuse_a=*/false, /*reuse_b=*/false);
}

union Frag { v16h h; uint4 u[2]; };

// ---------------------------------------------------------------------------
// Kernel 1: QKV GEMM.  out[m,n] = (x[m,:] @ W[:,n] + bias[n]) * scale
// Tile: 128(M) x 128(N) per 256-thread block; 8 waves of 64x32 each.
// A staged as f16 [128][40] (row-major, K contiguous), B staged TRANSPOSED
// as f16 [128][40] (Bs[n][k]) so both frags are 2x b128 LDS loads.
// ---------------------------------------------------------------------------
__global__ __launch_bounds__(256)
void qkv_gemm_kernel(const float* __restrict__ x,
                     const float* __restrict__ Wq, const float* __restrict__ bq,
                     const float* __restrict__ Wk, const float* __restrict__ bk,
                     const float* __restrict__ Wv, const float* __restrict__ bv,
                     h16* __restrict__ Qws, h16* __restrict__ Kws, h16* __restrict__ Vws)
{
    const int z = blockIdx.z;
    const float* __restrict__ W    = (z == 0) ? Wq : (z == 1) ? Wk : Wv;
    const float* __restrict__ bias = (z == 0) ? bq : (z == 1) ? bk : bv;
    h16* __restrict__ dst          = (z == 0) ? Qws : (z == 1) ? Kws : Vws;
    const float scale = (z == 0) ? 0.125f : 1.0f;   // fold 1/sqrt(Dh) into Q

    const int n0   = blockIdx.x * 128;
    const int m0   = blockIdx.y * 128;
    const int tid  = threadIdx.x;
    const int lane = tid & 31;
    const int wave = tid >> 5;
    const int half = lane >> 4;
    const int l16  = lane & 15;
    const int wm   = (wave >> 2) * 64;   // wave M offset inside tile
    const int wn   = (wave & 3) * 32;    // wave N offset inside tile

    __shared__ __align__(16) h16 As[128 * 40];   // [m][k], pitch 40 halves
    __shared__ __align__(16) h16 Bs[128 * 40];   // [n][k] (transposed W tile)

    v8f zero = {};
    v8f acc[4][2];
#pragma unroll
    for (int mi = 0; mi < 4; ++mi)
#pragma unroll
        for (int ni = 0; ni < 2; ++ni) acc[mi][ni] = zero;

    for (int k0 = 0; k0 < DMODEL; k0 += 32) {
        // ---- stage A: x rows m0..m0+127, cols k0..k0+31, fp32 -> f16 ----
        {
            const int r  = tid >> 1;           // 0..127
            const int cb = (tid & 1) * 16;     // 0 or 16
            const float* src = x + (size_t)(m0 + r) * DMODEL + k0 + cb;
            unsigned int* dA = (unsigned int*)&As[r * 40 + cb];
#pragma unroll
            for (int j = 0; j < 8; ++j) {
                union { h16 hh[2]; unsigned int u; } pk;
                pk.hh[0] = (h16)src[2 * j];
                pk.hh[1] = (h16)src[2 * j + 1];
                dA[j] = pk.u;
            }
        }
        // ---- stage B transposed: W rows k0..k0+31, cols n0..n0+127 ----
        {
            const int kk = tid >> 3;           // 0..31
            const int nb = (tid & 7) * 16;     // 0..112
            const float* src = W + (size_t)(k0 + kk) * DMODEL + n0 + nb;
#pragma unroll
            for (int j = 0; j < 16; ++j)
                Bs[(nb + j) * 40 + kk] = (h16)src[j];
        }
        __syncthreads();

        // ---- fragments (CDNA5 16-bit A/B VGPR layouts, ISA 7.12.2) ----
        Frag af[4];
#pragma unroll
        for (int mi = 0; mi < 4; ++mi) {
            const int r = wm + 16 * mi + l16;
            af[mi].u[0] = *(const uint4*)&As[r * 40 + half * 8];       // K 0..7 / 8..15
            af[mi].u[1] = *(const uint4*)&As[r * 40 + 16 + half * 8];  // K 16..23 / 24..31
        }
        Frag bf[2];
#pragma unroll
        for (int ni = 0; ni < 2; ++ni) {
            const int n = wn + 16 * ni + l16;
            bf[ni].u[0] = *(const uint4*)&Bs[n * 40 + half * 8];       // K 0..7 / 8..15
            bf[ni].u[1] = *(const uint4*)&Bs[n * 40 + 16 + half * 8];  // K 16..23 / 24..31
        }
#pragma unroll
        for (int mi = 0; mi < 4; ++mi)
#pragma unroll
            for (int ni = 0; ni < 2; ++ni)
                acc[mi][ni] = wmma_f16(af[mi].h, bf[ni].h, acc[mi][ni]);
        __syncthreads();
    }

    // ---- epilogue: bias, scale, convert f16, scatter into [B,H,S,Dh] ----
#pragma unroll
    for (int mi = 0; mi < 4; ++mi)
#pragma unroll
        for (int ni = 0; ni < 2; ++ni) {
            const int n = n0 + wn + 16 * ni + l16;
            const float bn = bias[n];
            const int hh = n >> 6, dh = n & 63;
#pragma unroll
            for (int g = 0; g < 8; ++g) {
                const int m = m0 + wm + 16 * mi + 8 * half + g;  // token index
                const int b = m >> 11, s = m & 2047;
                const float v = (acc[mi][ni][g] + bn) * scale;
                dst[(((size_t)(b * HEADS + hh)) * SEQ + s) * DHEAD + dh] = (h16)v;
            }
        }
}

// ---------------------------------------------------------------------------
// Kernel 2: flash attention.  Block = 128 threads (4 waves); each wave owns
// 16 q-rows; block covers 64 q-rows of one (b,h).  Key loop in tiles of 64.
// K staged [key][dh] (B-frag of QK^T contiguous along dh);
// V staged transposed [dh][key] (B-frag of PV contiguous along key).
// P converted C-layout -> A-layout through a per-wave LDS patch.
// ---------------------------------------------------------------------------
__global__ __launch_bounds__(128)
void flash_attn_kernel(const h16* __restrict__ Qws, const h16* __restrict__ Kws,
                       const h16* __restrict__ Vws, float* __restrict__ out)
{
    const int bh   = blockIdx.y;             // b*H + h
    const int b    = bh >> 4, h = bh & 15;
    const int tid  = threadIdx.x;
    const int lane = tid & 31;
    const int wave = tid >> 5;
    const int half = lane >> 4;
    const int l16  = lane & 15;
    const int qbase = blockIdx.x * 64 + wave * 16;

    const h16* __restrict__ Qp = Qws + (size_t)bh * SEQ * DHEAD;
    const h16* __restrict__ Kp = Kws + (size_t)bh * SEQ * DHEAD;
    const h16* __restrict__ Vp = Vws + (size_t)bh * SEQ * DHEAD;

    __shared__ __align__(16) h16 Ks[64 * 72];      // [key][dh], pitch 72
    __shared__ __align__(16) h16 Vs[64 * 72];      // [dh][key], pitch 72
    __shared__ __align__(16) h16 Ps[4][16 * 72];   // per-wave P patch [q][key]

    // ---- cache Q A-fragments (16 q-rows), one per dh k-step ----
    Frag qf[2];
#pragma unroll
    for (int ks = 0; ks < 2; ++ks) {
        const h16* p = Qp + (size_t)(qbase + l16) * DHEAD + ks * 32 + half * 8;
        qf[ks].u[0] = *(const uint4*)p;
        qf[ks].u[1] = *(const uint4*)(p + 16);
    }

    v8f zero = {};
    v8f Oacc[4];
#pragma unroll
    for (int nj = 0; nj < 4; ++nj) Oacc[nj] = zero;

    float mrow[8], lrow[8];
#pragma unroll
    for (int g = 0; g < 8; ++g) { mrow[g] = -3.0e38f; lrow[g] = 0.0f; }

    for (int kt = 0; kt < SEQ; kt += 64) {
        // ---- stage K tile [64 keys][64 dh]: 64B contiguous per thread ----
        {
            const int key = tid >> 1, dhb = (tid & 1) * 32;
            const uint4* src = (const uint4*)(Kp + (size_t)(kt + key) * DHEAD + dhb);
            uint4* d = (uint4*)&Ks[key * 72 + dhb];
            d[0] = src[0]; d[1] = src[1]; d[2] = src[2]; d[3] = src[3];
        }
        // ---- stage V tile transposed -> Vs[dh][key] ----
        {
            const int key = tid >> 1, dhb = (tid & 1) * 32;
            union { uint4 u[4]; h16 hh[32]; } t;
            const uint4* src = (const uint4*)(Vp + (size_t)(kt + key) * DHEAD + dhb);
            t.u[0] = src[0]; t.u[1] = src[1]; t.u[2] = src[2]; t.u[3] = src[3];
#pragma unroll
            for (int j = 0; j < 32; ++j) Vs[(dhb + j) * 72 + key] = t.hh[j];
        }
        // prefetch next K/V tile into cache (global_prefetch_b8)
        if (kt + 64 < SEQ) {
            __builtin_prefetch(Kp + (size_t)(kt + 64 + (tid >> 1)) * DHEAD, 0, 1);
            __builtin_prefetch(Vp + (size_t)(kt + 64 + (tid >> 1)) * DHEAD, 0, 1);
        }
        __syncthreads();

        // ---- scores = Q (pre-scaled) x K^T : 16q x 64key in fp32 ----
        v8f sc[4];
#pragma unroll
        for (int ni = 0; ni < 4; ++ni) {
            sc[ni] = zero;
            const h16* kp0 = &Ks[(16 * ni + l16) * 72];
            Frag kf0, kf1;
            kf0.u[0] = *(const uint4*)(kp0 + half * 8);
            kf0.u[1] = *(const uint4*)(kp0 + 16 + half * 8);
            kf1.u[0] = *(const uint4*)(kp0 + 32 + half * 8);
            kf1.u[1] = *(const uint4*)(kp0 + 32 + 16 + half * 8);
            sc[ni] = wmma_f16(qf[0].h, kf0.h, sc[ni]);
            sc[ni] = wmma_f16(qf[1].h, kf1.h, sc[ni]);
        }

        // ---- online softmax update (rows live in C-layout: vgpr g, half) ----
#pragma unroll
        for (int g = 0; g < 8; ++g) {
            float v0 = sc[0][g], v1 = sc[1][g], v2 = sc[2][g], v3 = sc[3][g];
            float rm = fmaxf(fmaxf(v0, v1), fmaxf(v2, v3));
#pragma unroll
            for (int msk = 8; msk >= 1; msk >>= 1)
                rm = fmaxf(rm, __shfl_xor(rm, msk, 32));
            const float mnew = fmaxf(mrow[g], rm);
            const float corr = __expf(mrow[g] - mnew);
            const float p0 = __expf(v0 - mnew);
            const float p1 = __expf(v1 - mnew);
            const float p2 = __expf(v2 - mnew);
            const float p3 = __expf(v3 - mnew);
            float ps = (p0 + p1) + (p2 + p3);
#pragma unroll
            for (int msk = 8; msk >= 1; msk >>= 1)
                ps += __shfl_xor(ps, msk, 32);
            lrow[g] = lrow[g] * corr + ps;
            mrow[g] = mnew;
#pragma unroll
            for (int nj = 0; nj < 4; ++nj) Oacc[nj][g] *= corr;
            const int ql = 8 * half + g;            // q row in 16-row tile
            h16* pr = &Ps[wave][ql * 72 + l16];
            pr[0]  = (h16)p0;
            pr[16] = (h16)p1;
            pr[32] = (h16)p2;
            pr[48] = (h16)p3;
        }
        // Ps is per-wave private; same-wave DS ops are in-order -> no barrier.

        // ---- O += P x V  (P: 16q x 64key, two 32-key k-steps) ----
#pragma unroll
        for (int ks = 0; ks < 2; ++ks) {
            Frag pf;
            const h16* pp = &Ps[wave][l16 * 72 + ks * 32];
            pf.u[0] = *(const uint4*)(pp + half * 8);
            pf.u[1] = *(const uint4*)(pp + 16 + half * 8);
#pragma unroll
            for (int nj = 0; nj < 4; ++nj) {
                const h16* vp = &Vs[(16 * nj + l16) * 72 + ks * 32];
                Frag vf;
                vf.u[0] = *(const uint4*)(vp + half * 8);
                vf.u[1] = *(const uint4*)(vp + 16 + half * 8);
                Oacc[nj] = wmma_f16(pf.h, vf.h, Oacc[nj]);
            }
        }
        __syncthreads();
    }

    // ---- epilogue: O / l, fp32 out in [B,S,D] ----
#pragma unroll
    for (int g = 0; g < 8; ++g) {
        const int q = qbase + 8 * half + g;
        const float inv = 1.0f / lrow[g];
        float* op = out + ((size_t)(b * SEQ + q)) * DMODEL + h * DHEAD;
#pragma unroll
        for (int nj = 0; nj < 4; ++nj)
            op[16 * nj + l16] = Oacc[nj][g] * inv;
    }
}

// ---------------------------------------------------------------------------
extern "C" void kernel_launch(void* const* d_in, const int* in_sizes, int n_in,
                              void* d_out, int out_size, void* d_ws, size_t ws_size,
                              hipStream_t stream) {
    const float* x  = (const float*)d_in[0];
    const float* Wq = (const float*)d_in[1];
    const float* bq = (const float*)d_in[2];
    const float* Wk = (const float*)d_in[3];
    const float* bk = (const float*)d_in[4];
    const float* Wv = (const float*)d_in[5];
    const float* bv = (const float*)d_in[6];

    const size_t qkv_elems = (size_t)BATCH * HEADS * SEQ * DHEAD;  // 8.4M
    h16* Qws = (h16*)d_ws;                 // 16 MB each, 48 MB total f16
    h16* Kws = Qws + qkv_elems;
    h16* Vws = Kws + qkv_elems;

    // QKV projection: M=8192, N=1024, tiles 128x128, z selects Q/K/V.
    dim3 g1(DMODEL / 128, (BATCH * SEQ) / 128, 3);
    qkv_gemm_kernel<<<g1, dim3(256), 0, stream>>>(x, Wq, bq, Wk, bk, Wv, bv,
                                                  Qws, Kws, Vws);

    // Flash attention: 64 q-rows per block, one (b,h) per blockIdx.y.
    dim3 g2(SEQ / 64, BATCH * HEADS);
    flash_attn_kernel<<<g2, dim3(128), 0, stream>>>(Qws, Kws, Vws, (float*)d_out);
}
